// CausalSelfAttention_1236950581811
// MI455X (gfx1250) — compile-verified, matrix operand order checked by reading, not run on hardware
//
#include <hip/hip_runtime.h>

typedef __attribute__((ext_vector_type(16))) __bf16 v16bf;
typedef __attribute__((ext_vector_type(8)))  __bf16 v8bf;
typedef __attribute__((ext_vector_type(4)))  __bf16 v4bf;
typedef __attribute__((ext_vector_type(8)))  float  v8f;
typedef __attribute__((ext_vector_type(4)))  int    v4i;

#define B_ 4
#define T_ 2048
#define C_ 1024
#define H_ 16
#define D_ 64
#define M_ (B_ * T_)   // 8192 rows

// ---------------------------------------------------------------------------
// CDNA5 async global->LDS copy (ASYNCcnt-tracked, no VGPR staging).
// Probe result: builtin takes (v4i AS1*, v4i AS3*, imm offset, imm cpol).
// Guarded by __has_builtin; fallback is a plain copy with identical control
// flow (still double-buffered).
// ---------------------------------------------------------------------------
#define HAS_ASYNC_LDS __has_builtin(__builtin_amdgcn_global_load_async_to_lds_b128)

#if HAS_ASYNC_LDS
__device__ __forceinline__ void async_cp16(void* lds, const void* g) {
  __builtin_amdgcn_global_load_async_to_lds_b128(
      (__attribute__((address_space(1))) v4i*)(g),
      (__attribute__((address_space(3))) v4i*)(lds), 0, 0);
}
#if __has_builtin(__builtin_amdgcn_s_wait_asynccnt)
#define WAIT_ASYNC(n) __builtin_amdgcn_s_wait_asynccnt(n)
#else
#define WAIT_ASYNC(n) asm volatile("s_wait_asynccnt %0" ::"i"(n) : "memory")
#endif
#else
#define WAIT_ASYNC(n) ((void)0)
#endif

// ---------------------------------------------------------------------------
// Fragment loader for V_WMMA_F32_16X16X32_BF16 (two ds_load_b128 per lane).
// A-matrix 16x32: lanes 0-15 hold K=0..7 & 16..23, lanes 16-31 K=8..15 &
// 24..31 -> two contiguous 16B runs at byte offsets klo*2 and klo*2+32 in a
// row-major [16][stride] tile. B of (x @ W^T) uses identical addressing.
// ---------------------------------------------------------------------------
__device__ __forceinline__ v16bf load_frag(const __bf16* base, int stride) {
  int lane = threadIdx.x & 31;
  int r    = lane & 15;
  int klo  = (lane & 16) >> 1;          // 0 or 8
  const __bf16* p = base + r * stride + klo;
  v8bf lo = *(const v8bf*)(p);
  v8bf hi = *(const v8bf*)(p + 16);
  v16bf f;
#pragma unroll
  for (int i = 0; i < 8; ++i) { f[i] = lo[i]; f[i + 8] = hi[i]; }
  return f;
}

__device__ __forceinline__ v8f wmma_bf16(v16bf a, v16bf b, v8f c) {
  return __builtin_amdgcn_wmma_f32_16x16x32_bf16(false, a, false, b,
                                                 (short)0, c, false, false);
}

// ---------------------------------------------------------------------------
// fp32 -> bf16 conversion, vectorized (float4 in, 4x bf16 packed out).
// ---------------------------------------------------------------------------
__global__ void cvt_f32_bf16(const float* __restrict__ s,
                             __bf16* __restrict__ d, int n4) {
  int i = blockIdx.x * blockDim.x + threadIdx.x;
  int stride = gridDim.x * blockDim.x;
  for (; i < n4; i += stride) {
    float4 v = ((const float4*)s)[i];
    v4bf o;
    o[0] = (__bf16)v.x; o[1] = (__bf16)v.y;
    o[2] = (__bf16)v.z; o[3] = (__bf16)v.w;
    ((v4bf*)d)[i] = o;
  }
}

// ---------------------------------------------------------------------------
// Shared GEMM core: 128x128 block tile, 8 waves of 32x64, K-chunk 32,
// double-buffered LDS fed by async-to-LDS. acc[2][4] per wave (8 wmma/chunk).
// ---------------------------------------------------------------------------
struct GemmTile {
  v8f acc[2][4];
  int mw, nw;          // wave offsets inside the 128x128 tile
};

__device__ __forceinline__ void gemm_core_128(
    const __bf16* __restrict__ Ag, const __bf16* __restrict__ Bg,
    int m0, int n0,
    __bf16 (*As)[128][32], __bf16 (*Bs)[128][32], GemmTile& gt) {
  int tid = threadIdx.x;
  int w = tid >> 5;
  gt.mw = (w & 3) * 32;                 // 0,32,64,96
  gt.nw = (w >> 2) * 64;                // 0 or 64
#pragma unroll
  for (int i = 0; i < 2; ++i)
#pragma unroll
    for (int j = 0; j < 4; ++j) gt.acc[i][j] = (v8f){};

  int lr = tid >> 1;                    // 0..127
  int lhalf = (tid & 1) * 16;           // 0 or 16

  auto issue = [&](int kc, int buf) {
    const __bf16* ga = &Ag[(size_t)(m0 + lr) * C_ + kc + lhalf];
    const __bf16* gb = &Bg[(size_t)(n0 + lr) * C_ + kc + lhalf];
#if HAS_ASYNC_LDS
    async_cp16(&As[buf][lr][lhalf],     ga);
    async_cp16(&As[buf][lr][lhalf + 8], ga + 8);
    async_cp16(&Bs[buf][lr][lhalf],     gb);
    async_cp16(&Bs[buf][lr][lhalf + 8], gb + 8);
#else
    *(v8bf*)&As[buf][lr][lhalf]     = *(const v8bf*)ga;
    *(v8bf*)&As[buf][lr][lhalf + 8] = *(const v8bf*)(ga + 8);
    *(v8bf*)&Bs[buf][lr][lhalf]     = *(const v8bf*)gb;
    *(v8bf*)&Bs[buf][lr][lhalf + 8] = *(const v8bf*)(gb + 8);
#endif
  };

  issue(0, 0);
  for (int kc = 0; kc < C_; kc += 32) {
    int cur = (kc >> 5) & 1;
    if (kc + 32 < C_) {                 // prefetch next chunk into other buf
      issue(kc + 32, cur ^ 1);
      WAIT_ASYNC(4);                    // in-order: current chunk has landed
    } else {
      WAIT_ASYNC(0);
    }
    __syncthreads();
    v16bf a0 = load_frag(&As[cur][gt.mw][0], 32);
    v16bf a1 = load_frag(&As[cur][gt.mw + 16][0], 32);
    v16bf b0 = load_frag(&Bs[cur][gt.nw][0], 32);
    v16bf b1 = load_frag(&Bs[cur][gt.nw + 16][0], 32);
    v16bf b2 = load_frag(&Bs[cur][gt.nw + 32][0], 32);
    v16bf b3 = load_frag(&Bs[cur][gt.nw + 48][0], 32);
    gt.acc[0][0] = wmma_bf16(a0, b0, gt.acc[0][0]);
    gt.acc[0][1] = wmma_bf16(a0, b1, gt.acc[0][1]);
    gt.acc[0][2] = wmma_bf16(a0, b2, gt.acc[0][2]);
    gt.acc[0][3] = wmma_bf16(a0, b3, gt.acc[0][3]);
    gt.acc[1][0] = wmma_bf16(a1, b0, gt.acc[1][0]);
    gt.acc[1][1] = wmma_bf16(a1, b1, gt.acc[1][1]);
    gt.acc[1][2] = wmma_bf16(a1, b2, gt.acc[1][2]);
    gt.acc[1][3] = wmma_bf16(a1, b3, gt.acc[1][3]);
    __syncthreads();                    // all reads of 'cur' done before reuse
  }
}

// ---------------------------------------------------------------------------
// QKV projection: out[b,h,t,d] = x @ W^T + bias (one z-slice per matrix)
// ---------------------------------------------------------------------------
__global__ __launch_bounds__(256) void qkv_gemm(
    const __bf16* __restrict__ xbf,
    const __bf16* __restrict__ Wqb, const __bf16* __restrict__ Wkb,
    const __bf16* __restrict__ Wvb,
    const float* __restrict__ bq, const float* __restrict__ bk,
    const float* __restrict__ bv,
    __bf16* __restrict__ Qb, __bf16* __restrict__ Kb,
    __bf16* __restrict__ Vb) {
  const __bf16* W; const float* bias; __bf16* out;
  int z = blockIdx.z;
  if (z == 0)      { W = Wqb; bias = bq; out = Qb; }
  else if (z == 1) { W = Wkb; bias = bk; out = Kb; }
  else             { W = Wvb; bias = bv; out = Vb; }

  __shared__ __align__(16) __bf16 As[2][128][32];
  __shared__ __align__(16) __bf16 Bs[2][128][32];
  int m0 = blockIdx.y * 128;
  int n0 = blockIdx.x * 128;

  GemmTile gt;
  gemm_core_128(xbf, W, m0, n0, As, Bs, gt);

  int lane = threadIdx.x & 31;
  int g = lane >> 4, col = lane & 15;
#pragma unroll
  for (int i = 0; i < 2; ++i)
#pragma unroll
    for (int j = 0; j < 4; ++j)
#pragma unroll
      for (int v = 0; v < 8; ++v) {
        int m = m0 + gt.mw + 16 * i + v + 8 * g;
        int b = m >> 11, t = m & (T_ - 1);
        int n = n0 + gt.nw + 16 * j + col;
        out[(((size_t)(b * H_ + (n >> 6)) * T_) + t) * D_ + (n & 63)] =
            (__bf16)(gt.acc[i][j][v] + bias[n]);
      }
}

// ---------------------------------------------------------------------------
// Output projection: d_out = y @ Wp^T + bp (fp32 epilogue)
// ---------------------------------------------------------------------------
__global__ __launch_bounds__(256) void proj_gemm(
    const __bf16* __restrict__ ybf, const __bf16* __restrict__ Wpb,
    const float* __restrict__ bp, float* __restrict__ out) {
  __shared__ __align__(16) __bf16 As[2][128][32];
  __shared__ __align__(16) __bf16 Bs[2][128][32];
  int m0 = blockIdx.y * 128;
  int n0 = blockIdx.x * 128;

  GemmTile gt;
  gemm_core_128(ybf, Wpb, m0, n0, As, Bs, gt);

  int lane = threadIdx.x & 31;
  int g = lane >> 4, col = lane & 15;
#pragma unroll
  for (int i = 0; i < 2; ++i)
#pragma unroll
    for (int j = 0; j < 4; ++j)
#pragma unroll
      for (int v = 0; v < 8; ++v) {
        int m = m0 + gt.mw + 16 * i + v + 8 * g;
        int n = n0 + gt.nw + 16 * j + col;
        out[(size_t)m * C_ + n] = gt.acc[i][j][v] + bp[n];
      }
}

// ---------------------------------------------------------------------------
// Flash attention: one (b,h) x 128 q-rows per 256-thread block (8 waves x 16
// rows). Keys streamed in 32-key chunks (K via async-to-LDS, V transposed
// through regs); online softmax; P@V via LDS V^T. y written [B,T,C] bf16.
// ---------------------------------------------------------------------------
__global__ __launch_bounds__(256) void flash_attn(
    const __bf16* __restrict__ Qb, const __bf16* __restrict__ Kb,
    const __bf16* __restrict__ Vb, __bf16* __restrict__ ybf) {
  __shared__ __align__(16) __bf16 Ks[32][64];
  __shared__ __align__(16) __bf16 Vt[64][32];
  __shared__ __align__(16) __bf16 Ps[8][16][32];

  int bh = blockIdx.y;
  int qbase = blockIdx.x * 128;
  const __bf16* Qh = Qb + (size_t)bh * T_ * D_;
  const __bf16* Kh = Kb + (size_t)bh * T_ * D_;
  const __bf16* Vh = Vb + (size_t)bh * T_ * D_;

  int tid = threadIdx.x, w = tid >> 5, lane = tid & 31;
  int qrow = qbase + w * 16;
  int g = lane >> 4, col = lane & 15;
  int kr = tid >> 3;                    // 0..31  (K/V row in chunk)
  int kc8 = (tid & 7) * 8;              // 0..56  (d offset, 8 elements)

  // Q fragments stay in registers for the whole kernel (16x64 = 2 frags)
  v16bf qa0 = load_frag(Qh + (size_t)qrow * D_, D_);
  v16bf qa1 = load_frag(Qh + (size_t)qrow * D_ + 32, D_);

  v8f O0 = {}, O1 = {}, O2 = {}, O3 = {};
  float mrow_[8], lrow_[8];
#pragma unroll
  for (int v = 0; v < 8; ++v) { mrow_[v] = -1e30f; lrow_[v] = 0.f; }

  const float scale = 0.125f;           // 1/sqrt(64)
  const float L2E = 1.44269504f;
  int nchunks = qbase / 32 + 4;         // keys 0 .. qbase+127

  for (int jc = 0; jc < nchunks; ++jc) {
    int jb = jc * 32;
    // K chunk (32x64) -> LDS, one async b128 per thread
#if HAS_ASYNC_LDS
    async_cp16(&Ks[kr][kc8], &Kh[(size_t)(jb + kr) * D_ + kc8]);
#else
    *(v8bf*)&Ks[kr][kc8] = *(const v8bf*)&Kh[(size_t)(jb + kr) * D_ + kc8];
#endif
    // V chunk -> LDS transposed (Vt[d][j]) so P@V reuses the fragment loader
    v8bf vv = *(const v8bf*)&Vh[(size_t)(jb + kr) * D_ + kc8];
#pragma unroll
    for (int i = 0; i < 8; ++i) Vt[kc8 + i][kr] = vv[i];
    WAIT_ASYNC(0);
    __syncthreads();

    // S = Q K^T (16 q-rows x 32 keys), K-dim 64 -> 2 wmma per score tile
    v8f s0 = {}, s1 = {};
    s0 = wmma_bf16(qa0, load_frag(&Ks[0][0], 64), s0);
    s0 = wmma_bf16(qa1, load_frag(&Ks[0][32], 64), s0);
    s1 = wmma_bf16(qa0, load_frag(&Ks[16][0], 64), s1);
    s1 = wmma_bf16(qa1, load_frag(&Ks[16][32], 64), s1);

    // causal mask + scale + row max (rows live in 16-lane halves)
    float rmax[8];
#pragma unroll
    for (int v = 0; v < 8; ++v) {
      int q = qrow + v + 8 * g;
      s0[v] = (jb + col <= q)      ? s0[v] * scale : -1e30f;
      s1[v] = (jb + 16 + col <= q) ? s1[v] * scale : -1e30f;
      float r = fmaxf(s0[v], s1[v]);
      r = fmaxf(r, __shfl_xor(r, 1, 32));
      r = fmaxf(r, __shfl_xor(r, 2, 32));
      r = fmaxf(r, __shfl_xor(r, 4, 32));
      r = fmaxf(r, __shfl_xor(r, 8, 32));
      rmax[v] = r;
    }

    __bf16* ps = &Ps[w][0][0];
    float sc[8];
#pragma unroll
    for (int v = 0; v < 8; ++v) {
      float mn = fmaxf(mrow_[v], rmax[v]);
      sc[v] = exp2f((mrow_[v] - mn) * L2E);
      mrow_[v] = mn;
      float p0 = exp2f((s0[v] - mn) * L2E);
      float p1 = exp2f((s1[v] - mn) * L2E);
      float rs = p0 + p1;
      rs += __shfl_xor(rs, 1, 32);
      rs += __shfl_xor(rs, 2, 32);
      rs += __shfl_xor(rs, 4, 32);
      rs += __shfl_xor(rs, 8, 32);
      lrow_[v] = lrow_[v] * sc[v] + rs;
      int row = v + 8 * g;
      ps[row * 32 + col]      = (__bf16)p0;   // C-layout -> A-layout via LDS
      ps[row * 32 + 16 + col] = (__bf16)p1;
      O0[v] *= sc[v]; O1[v] *= sc[v]; O2[v] *= sc[v]; O3[v] *= sc[v];
    }
    asm volatile("s_wait_dscnt 0" ::: "memory");  // P stores before frag load

    v16bf pf = load_frag(ps, 32);
    O0 = wmma_bf16(pf, load_frag(&Vt[0][0], 32),  O0);
    O1 = wmma_bf16(pf, load_frag(&Vt[16][0], 32), O1);
    O2 = wmma_bf16(pf, load_frag(&Vt[32][0], 32), O2);
    O3 = wmma_bf16(pf, load_frag(&Vt[48][0], 32), O3);
    __syncthreads();                    // Ks/Vt reuse next chunk
  }

  int b = bh >> 4, h = bh & 15;
#pragma unroll
  for (int v = 0; v < 8; ++v) {
    float inv = 1.0f / lrow_[v];
    int t = qrow + v + 8 * g;
    size_t base = ((size_t)(b * T_ + t)) * C_ + h * D_;
    ybf[base + col]      = (__bf16)(O0[v] * inv);
    ybf[base + 16 + col] = (__bf16)(O1[v] * inv);
    ybf[base + 32 + col] = (__bf16)(O2[v] * inv);
    ybf[base + 48 + col] = (__bf16)(O3[v] * inv);
  }
}

// ---------------------------------------------------------------------------
extern "C" void kernel_launch(void* const* d_in, const int* in_sizes, int n_in,
                              void* d_out, int out_size, void* d_ws,
                              size_t ws_size, hipStream_t stream) {
  const float* x  = (const float*)d_in[0];
  const float* Wq = (const float*)d_in[1];
  const float* bq = (const float*)d_in[2];
  const float* Wk = (const float*)d_in[3];
  const float* bk = (const float*)d_in[4];
  const float* Wv = (const float*)d_in[5];
  const float* bv = (const float*)d_in[6];
  const float* Wp = (const float*)d_in[7];
  const float* bp = (const float*)d_in[8];
  float* out = (float*)d_out;

  char* ws = (char*)d_ws;
  size_t off = 0;
  const size_t NX = (size_t)M_ * C_;    // 8M elements
  const size_t NW = (size_t)C_ * C_;    // 1M elements
  __bf16* xbf = (__bf16*)(ws + off); off += NX * 2;
  __bf16* Wqb = (__bf16*)(ws + off); off += NW * 2;
  __bf16* Wkb = (__bf16*)(ws + off); off += NW * 2;
  __bf16* Wvb = (__bf16*)(ws + off); off += NW * 2;
  __bf16* Wpb = (__bf16*)(ws + off); off += NW * 2;
  __bf16* Qb  = (__bf16*)(ws + off); off += NX * 2;
  __bf16* Kb  = (__bf16*)(ws + off); off += NX * 2;
  __bf16* Vb  = (__bf16*)(ws + off); off += NX * 2;
  __bf16* ybf = (__bf16*)(ws + off); off += NX * 2;   // ~88 MB total

  // 1) bf16 conversion (vectorized x4)
  cvt_f32_bf16<<<1024, 256, 0, stream>>>(x,  xbf, (int)(NX / 4));
  cvt_f32_bf16<<<512, 256, 0, stream>>>(Wq, Wqb, (int)(NW / 4));
  cvt_f32_bf16<<<512, 256, 0, stream>>>(Wk, Wkb, (int)(NW / 4));
  cvt_f32_bf16<<<512, 256, 0, stream>>>(Wv, Wvb, (int)(NW / 4));
  cvt_f32_bf16<<<512, 256, 0, stream>>>(Wp, Wpb, (int)(NW / 4));

  // 2) QKV projections (z = 0,1,2)
  qkv_gemm<<<dim3(C_ / 128, M_ / 128, 3), 256, 0, stream>>>(
      xbf, Wqb, Wkb, Wvb, bq, bk, bv, Qb, Kb, Vb);

  // 3) causal flash attention
  flash_attn<<<dim3(T_ / 128, B_ * H_), 256, 0, stream>>>(Qb, Kb, Vb, ybf);

  // 4) output projection
  proj_gemm<<<dim3(C_ / 128, M_ / 128), 256, 0, stream>>>(ybf, Wpb, bp, out);
}
